// GPT2Attention_9328668966923
// MI455X (gfx1250) — compile-verified, matrix operand order checked by reading, not run on hardware
//
#include <hip/hip_runtime.h>
#include <hip/hip_bf16.h>

// ---------------------------------------------------------------------------
// GPT-2 attention forward for gfx1250 (MI455X), bf16 WMMA + f32 accumulate.
// Double-buffered GEMM K-loop; fused softmax-normalize; NT stores on finals.
// ---------------------------------------------------------------------------

typedef __attribute__((ext_vector_type(16))) __bf16 v16bf;
typedef __attribute__((ext_vector_type(8)))  float  v8f;
typedef __attribute__((ext_vector_type(2)))  float  f32x2;   // native vec2 for NT stores

union FragBF { v16bf v; unsigned int u[8]; unsigned short h[16]; };
union FragF  { v8f   v; float f[8]; };

#define S_LEN   2048
#define D_EMB   1024
#define N_HEADS 16
#define HD      64
#define BH      32        // B * N_HEADS
#define M_ROWS  4096      // B * S
#define N3      3072      // 3 * D

__device__ __forceinline__ unsigned short f2bf(float x) {
    unsigned int u = __float_as_uint(x);
    unsigned int r = u + 0x7FFFu + ((u >> 16) & 1u);   // round-to-nearest-even
    return (unsigned short)(r >> 16);
}
__device__ __forceinline__ unsigned int pack_bf2(float a, float b) {
    return (unsigned int)f2bf(a) | ((unsigned int)f2bf(b) << 16);
}
// per-lane K offset for 16-bit A/B fragments (ISA 7.12.2), j = reg index 0..7
__device__ __forceinline__ int koff(int j, int half) {
    return (j < 4) ? (2 * j + 8 * half) : (16 + 2 * (j - 4) + 8 * half);
}
__device__ __forceinline__ void load_frag(FragBF& f, const unsigned short* base,
                                          int kc, int half) {
#pragma unroll
    for (int j = 0; j < 8; ++j)
        f.u[j] = *(const unsigned int*)(base + kc + koff(j, half));
}

// ---------------------------------------------------------------------------
// elementwise converts
// ---------------------------------------------------------------------------
__global__ void k_convert_x(const float* __restrict__ x,
                            unsigned short* __restrict__ xb, int n) {
    int i = blockIdx.x * blockDim.x + threadIdx.x;
    if (i < n) xb[i] = f2bf(x[i]);
}

// W [K,N] f32 -> Wt [N,K] bf16
__global__ void k_transpose_w(const float* __restrict__ W,
                              unsigned short* __restrict__ Wt, int K, int N) {
    int i = blockIdx.x * blockDim.x + threadIdx.x;
    if (i < K * N) {
        int n = i / K, k = i % K;
        Wt[i] = f2bf(W[(size_t)k * N + n]);
    }
}

// ---------------------------------------------------------------------------
// Generic GEMM: C = A(bf16,[M,K]) * Bt(bf16,[N,K])^T + bias(f32)
// Block: 256 thr = 8 waves, tile 64(M) x 128(N); wave tile 16x64.
// K-loop software-pipelined (double-buffered fragments), K % 64 == 0.
// ---------------------------------------------------------------------------
__global__ void k_gemm_bf16(const unsigned short* __restrict__ A,
                            const unsigned short* __restrict__ Bt,
                            const float* __restrict__ bias,
                            float* __restrict__ outF,
                            unsigned short* __restrict__ outH,
                            int Nn, int K) {
    int tid  = threadIdx.x;
    int lane = tid & 31, wave = tid >> 5;
    int half = lane >> 4, r16 = lane & 15;
    int wrow = wave & 3, wcol = wave >> 2;
    int mbase = blockIdx.y * 64 + wrow * 16;
    int nbase = blockIdx.x * 128 + wcol * 64;

    const unsigned short* arow = A + (size_t)(mbase + r16) * K;
    const unsigned short* bp[4];
#pragma unroll
    for (int g = 0; g < 4; ++g)
        bp[g] = Bt + (size_t)(nbase + g * 16 + r16) * K;

    FragF acc[4];
#pragma unroll
    for (int g = 0; g < 4; ++g)
#pragma unroll
        for (int j = 0; j < 8; ++j) acc[g].f[j] = 0.0f;

    FragBF aC, aN, bC[4], bN[4];
    load_frag(aC, arow, 0, half);
#pragma unroll
    for (int g = 0; g < 4; ++g) load_frag(bC[g], bp[g], 0, half);

    for (int kc = 0; kc < K; kc += 64) {
        // prefetch chunk kc+32 while wmma on kc
        load_frag(aN, arow, kc + 32, half);
#pragma unroll
        for (int g = 0; g < 4; ++g) load_frag(bN[g], bp[g], kc + 32, half);
#pragma unroll
        for (int g = 0; g < 4; ++g)
            acc[g].v = __builtin_amdgcn_wmma_f32_16x16x32_bf16(
                false, aC.v, false, bC[g].v, (short)0, acc[g].v, false, false);

        int k2 = kc + 64;
        if (k2 < K) {
            load_frag(aC, arow, k2, half);
#pragma unroll
            for (int g = 0; g < 4; ++g) load_frag(bC[g], bp[g], k2, half);
        }
#pragma unroll
        for (int g = 0; g < 4; ++g)
            acc[g].v = __builtin_amdgcn_wmma_f32_16x16x32_bf16(
                false, aN.v, false, bN[g].v, (short)0, acc[g].v, false, false);
    }

#pragma unroll
    for (int g = 0; g < 4; ++g) {
        int col = nbase + g * 16 + r16;
        float bs = bias[col];
#pragma unroll
        for (int j = 0; j < 8; ++j) {
            int row = mbase + j + 8 * half;
            float val = acc[g].f[j] + bs;
            if (outF)   // final output: never re-read -> non-temporal
                __builtin_nontemporal_store(val, &outF[(size_t)row * Nn + col]);
            if (outH)   // intermediate (re-read): default temporal policy
                outH[(size_t)row * Nn + col] = f2bf(val);
        }
    }
}

// ---------------------------------------------------------------------------
// qkv [M, 3072] bf16 -> q,k [BH,S,64] and v^T [BH,64,S]
// ---------------------------------------------------------------------------
__global__ void k_reshuffle(const unsigned short* __restrict__ qkv,
                            unsigned short* __restrict__ q,
                            unsigned short* __restrict__ k,
                            unsigned short* __restrict__ vt) {
    int i = blockIdx.x * blockDim.x + threadIdx.x;
    if (i >= M_ROWS * N3) return;
    int row  = i / N3;           // b*S + s
    int c    = i % N3;
    int part = c >> 10;
    int cd   = c & 1023;
    int h = cd >> 6, d = cd & 63;
    int b = row >> 11, s = row & 2047;
    int bh = b * N_HEADS + h;
    unsigned short val = qkv[i];
    if (part == 0)      q[((size_t)bh * S_LEN + s) * HD + d]  = val;
    else if (part == 1) k[((size_t)bh * S_LEN + s) * HD + d]  = val;
    else                vt[((size_t)bh * HD + d) * S_LEN + s] = val;
}

// ---------------------------------------------------------------------------
// Pass 1: scores = q @ k^T / 8, causal mask, write raw, reduce row-max.
// Block 128 thr = 4 waves; block covers (bh, 16 q-rows); wave = column tiles.
// ---------------------------------------------------------------------------
__global__ void k_scores(const unsigned short* __restrict__ q,
                         const unsigned short* __restrict__ kk,
                         float* __restrict__ attn,
                         float* __restrict__ rowmax) {
    __shared__ float lds[4 * 32 * 8];
    int bh = blockIdx.y, qt = blockIdx.x;
    int qbase = qt * 16;
    int tid = threadIdx.x, lane = tid & 31, wave = tid >> 5;
    int half = lane >> 4, r16 = lane & 15;

    const unsigned short* qrow = q + ((size_t)bh * S_LEN + qbase + r16) * HD;
    FragBF a0, a1;
    load_frag(a0, qrow, 0, half);
    load_frag(a1, qrow, 32, half);

    float rmax[8];
#pragma unroll
    for (int j = 0; j < 8; ++j) rmax[j] = -3.0e38f;

    float* arow = attn + (size_t)bh * S_LEN * S_LEN;
    int ntiles = qt + 1;   // causal: only tiles up to/including diagonal tile
    for (int ct = wave; ct < ntiles; ct += 4) {
        int cbase = ct * 16;
        const unsigned short* krow =
            kk + ((size_t)bh * S_LEN + cbase + r16) * HD;
        FragBF b0, b1;
        load_frag(b0, krow, 0, half);
        load_frag(b1, krow, 32, half);
        FragF acc;
#pragma unroll
        for (int j = 0; j < 8; ++j) acc.f[j] = 0.0f;
        acc.v = __builtin_amdgcn_wmma_f32_16x16x32_bf16(
            false, a0.v, false, b0.v, (short)0, acc.v, false, false);
        acc.v = __builtin_amdgcn_wmma_f32_16x16x32_bf16(
            false, a1.v, false, b1.v, (short)0, acc.v, false, false);

        int col = cbase + r16;
#pragma unroll
        for (int j = 0; j < 8; ++j) {
            int row = qbase + j + 8 * half;
            float v = acc.f[j] * 0.125f;          // 1/sqrt(64)
            if (col > row) v = -1.0e30f;          // causal mask
            arow[(size_t)row * S_LEN + col] = v;  // re-read by pass 2: RT
            rmax[j] = fmaxf(rmax[j], v);
        }
    }

#pragma unroll
    for (int j = 0; j < 8; ++j) lds[(wave * 32 + lane) * 8 + j] = rmax[j];
    __syncthreads();
    if (tid < 16) {
        int h2 = tid >> 3, j = tid & 7;
        float m = -3.0e38f;
        for (int w = 0; w < 4; ++w)
            for (int l = 0; l < 16; ++l)
                m = fmaxf(m, lds[(w * 32 + h2 * 16 + l) * 8 + j]);
        rowmax[(size_t)bh * S_LEN + qbase + tid] = m;
    }
}

// ---------------------------------------------------------------------------
// Pass 2 (fused softmax + p@v):
//   sweep 1: row-sum of exp(s - max) (chunks split across the 4 waves)
//   sweep 2: recompute exp (L2-hot), write FINAL normalized weights (NT),
//            feed normalized bf16 p into WMMA with v -> head output.
// Block 128 thr = 4 waves; each wave owns 16 of the 64 head-dim columns.
// ---------------------------------------------------------------------------
__global__ void k_softmax_av(const float* __restrict__ rowmax,
                             const unsigned short* __restrict__ vt,
                             float* __restrict__ attn,
                             unsigned short* __restrict__ attn_out) {
    __shared__ float lds_sum[128];
    __shared__ float lds_inv[16];
    int bh = blockIdx.y, qt = blockIdx.x;
    int qbase = qt * 16;
    int tid = threadIdx.x, lane = tid & 31, wave = tid >> 5;
    int half = lane >> 4, r16 = lane & 15;

    int validCols = qbase + 16;
    int nch = (validCols + 31) >> 5;
    float* arow = attn + (size_t)bh * S_LEN * S_LEN;
    int row = qbase + r16;                         // A-fragment row (lane%16)
    float rmax = rowmax[(size_t)bh * S_LEN + row];

    // ---- sweep 1: row sums (chunks distributed across waves) ----
    float rsum = 0.0f;
    for (int c0 = wave; c0 < nch; c0 += 4) {
        int kb = c0 * 32;
#pragma unroll
        for (int j = 0; j < 8; ++j) {
            int kidx = kb + koff(j, half);
            if (kidx < validCols) {
                const f32x2 sv =
                    *(const f32x2*)(arow + (size_t)row * S_LEN + kidx);
                rsum += __expf(sv.x - rmax) + __expf(sv.y - rmax);
            }
        }
    }
    lds_sum[wave * 32 + lane] = rsum;
    __syncthreads();
    if (tid < 16) {
        float s = 0.0f;
        for (int w = 0; w < 4; ++w)
            s += lds_sum[w * 32 + tid] + lds_sum[w * 32 + 16 + tid];
        lds_inv[tid] = 1.0f / s;
    }
    __syncthreads();
    float rinv = lds_inv[r16];

    // ---- sweep 2: normalized weights (NT store) + p@v via WMMA ----
    FragF acc;
#pragma unroll
    for (int j = 0; j < 8; ++j) acc.f[j] = 0.0f;
    const unsigned short* vcol =
        vt + ((size_t)bh * HD + wave * 16 + r16) * S_LEN;

    for (int c0 = 0; c0 < nch; ++c0) {
        int kb = c0 * 32;
        FragBF a;
#pragma unroll
        for (int j = 0; j < 8; ++j) {
            int kidx = kb + koff(j, half);
            float p0 = 0.0f, p1 = 0.0f;
            if (kidx < validCols) {
                const f32x2 sv =
                    *(const f32x2*)(arow + (size_t)row * S_LEN + kidx);
                p0 = __expf(sv.x - rmax) * rinv;
                p1 = __expf(sv.y - rmax) * rinv;
                if (wave == 0) {  // final attn weights: non-temporal
                    f32x2 pv;
                    pv.x = p0;
                    pv.y = p1;
                    __builtin_nontemporal_store(
                        pv, (f32x2*)(arow + (size_t)row * S_LEN + kidx));
                }
            }
            a.u[j] = pack_bf2(p0, p1);
        }
        FragBF b;
        load_frag(b, vcol, kb, half);
        acc.v = __builtin_amdgcn_wmma_f32_16x16x32_bf16(
            false, a.v, false, b.v, (short)0, acc.v, false, false);
    }

    int b = bh >> 4, h = bh & 15;
#pragma unroll
    for (int j = 0; j < 8; ++j) {
        int rloc = j + 8 * half;
        int s = qbase + rloc;
        int col = wave * 16 + r16;
        attn_out[((size_t)(b * S_LEN + s)) * D_EMB + h * HD + col] =
            f2bf(acc.f[j]);
    }
}

// ---------------------------------------------------------------------------
// Zero-fill the strictly-upper tile region of attn (never touched by pass1/2).
// ---------------------------------------------------------------------------
__global__ void k_zerofill(float* __restrict__ attn) {
    size_t i = (size_t)blockIdx.x * blockDim.x + threadIdx.x;
    if (i >= (size_t)BH * S_LEN * S_LEN) return;
    int c = (int)(i & 2047);
    int r = (int)((i >> 11) & 2047);
    if ((c >> 4) > (r >> 4))
        __builtin_nontemporal_store(0.0f, &attn[i]);
}

// ---------------------------------------------------------------------------
extern "C" void kernel_launch(void* const* d_in, const int* in_sizes, int n_in,
                              void* d_out, int out_size, void* d_ws, size_t ws_size,
                              hipStream_t stream) {
    const float* x      = (const float*)d_in[0];
    const float* W_attn = (const float*)d_in[1];
    const float* b_attn = (const float*)d_in[2];
    const float* W_proj = (const float*)d_in[3];
    const float* b_proj = (const float*)d_in[4];

    float* outF = (float*)d_out;                       // [2,2048,1024]
    float* attn = outF + (size_t)M_ROWS * D_EMB;       // [2,16,2048,2048]

    char* ws = (char*)d_ws;
    unsigned short* xb   = (unsigned short*)(ws);                  // 8 MB
    unsigned short* wat  = (unsigned short*)(ws + 8388608);        // 6 MB
    unsigned short* wpt  = (unsigned short*)(ws + 14680064);       // 2 MB
    unsigned short* qkvb = (unsigned short*)(ws + 16777216);       // 24 MB
    unsigned short* qh   = (unsigned short*)(ws + 41943040);       // 8 MB
    unsigned short* kh   = (unsigned short*)(ws + 50331648);       // 8 MB
    unsigned short* vth  = (unsigned short*)(ws + 58720256);       // 8 MB
    float*          rmax = (float*)(ws + 67108864);                // 256 KB
    unsigned short* aout = (unsigned short*)(ws + 67371008);       // 8 MB

    // 1) converts / transposes
    k_convert_x<<<(M_ROWS * D_EMB + 255) / 256, 256, 0, stream>>>(
        x, xb, M_ROWS * D_EMB);
    k_transpose_w<<<(D_EMB * N3 + 255) / 256, 256, 0, stream>>>(
        W_attn, wat, D_EMB, N3);
    k_transpose_w<<<(D_EMB * D_EMB + 255) / 256, 256, 0, stream>>>(
        W_proj, wpt, D_EMB, D_EMB);

    // 2) zero-fill strictly-upper attn region (independent of everything)
    k_zerofill<<<(unsigned)(((size_t)BH * S_LEN * S_LEN + 255) / 256), 256, 0,
                 stream>>>(attn);

    // 3) qkv = x @ W_attn + b_attn   (bf16 out)
    k_gemm_bf16<<<dim3(N3 / 128, M_ROWS / 64), 256, 0, stream>>>(
        xb, wat, b_attn, nullptr, qkvb, N3, D_EMB);

    // 4) head split / v transpose
    k_reshuffle<<<(M_ROWS * N3 + 255) / 256, 256, 0, stream>>>(
        qkvb, qh, kh, vth);

    // 5) raw causal scores + row max
    k_scores<<<dim3(S_LEN / 16, BH), 128, 0, stream>>>(qh, kh, attn, rmax);

    // 6) fused softmax-normalize + p@v (writes final attn weights + head out)
    k_softmax_av<<<dim3(S_LEN / 16, BH), 128, 0, stream>>>(
        rmax, vth, attn, aout);

    // 7) out = attn_out @ W_proj + b_proj  (f32 out, NT stores)
    k_gemm_bf16<<<dim3(D_EMB / 128, M_ROWS / 64), 256, 0, stream>>>(
        aout, wpt, b_proj, outF, nullptr, D_EMB, D_EMB);
}
